// Step1_Model_55284819034178
// MI455X (gfx1250) — compile-verified, matrix operand order checked by reading, not run on hardware
//
#include <hip/hip_runtime.h>
#include <hip/hip_bf16.h>
#include <math.h>

#define S_SEQ   241
#define NTOKN   240
#define DMODEL  128
#define DFFN    512
#define NEXP    8
#define NHEAD   4
#define DHEAD   32
#define NLAYER  2
#define NBATCH  128
#define SEGL    250
#define NFLAT   264                      // 33*8
#define NROWS   (NBATCH * S_SEQ)         // 30848 = 1928*16

typedef __attribute__((ext_vector_type(16))) _Float16 v16h;
typedef __attribute__((ext_vector_type(8)))  _Float16 v8h;
typedef __attribute__((ext_vector_type(8)))  float    v8f;
typedef __attribute__((ext_vector_type(4)))  unsigned int v4u;
typedef __attribute__((ext_vector_type(8)))  int v8i;
typedef __attribute__((ext_vector_type(4)))  int v4i;

// ---- WMMA helpers (CDNA5 wave32, 16x16x32 f16 -> f32) ----------------------
// 16-bit operand K mapping (ISA 7.12.2): lanes 0-15 hold K 0..7 / 16..23,
// lanes 16-31 hold K 8..15 / 24..31.  C/D: VGPR i = row i (lanes 0-15) or
// row 8+i (lanes 16-31), col = lane&15.

__device__ __forceinline__ v8f wmma16(v16h a, v16h b, v8f c) {
  return __builtin_amdgcn_wmma_f32_16x16x32_f16(false, a, false, b, (short)0, c,
                                                false, false);
}

__device__ __forceinline__ v16h pack16(v8h lo, v8h hi) {
  v16h r;
#pragma unroll
  for (int e = 0; e < 8; ++e) { r[e] = lo[e]; r[e + 8] = hi[e]; }
  return r;
}

// A tile, global f16 row-major (lane = row); two b128 loads
__device__ __forceinline__ v16h ldA_h(const _Float16* __restrict__ A, int lda,
                                      int mbase, int k0, int lane) {
  int m = mbase + (lane & 15);
  int kb = (lane & 16) ? 8 : 0;
  const _Float16* p = A + (size_t)m * lda + k0 + kb;
  return pack16(*(const v8h*)p, *(const v8h*)(p + 16));
}
// B(k,n) = W[n*ldw + k], W f16 row-major (N,K) (lane = col n); two b128 loads
__device__ __forceinline__ v16h ldB_hT(const _Float16* __restrict__ W, int ldw,
                                       int nbase, int k0, int lane) {
  int n = nbase + (lane & 15);
  int kb = (lane & 16) ? 8 : 0;
  const _Float16* p = W + (size_t)n * ldw + k0 + kb;
  return pack16(*(const v8h*)p, *(const v8h*)(p + 16));
}
// B(k,n) = W[n*ldw + k], W f32 row-major (N,K); four b128 loads + cvt
__device__ __forceinline__ v16h ldB_wT4(const float* __restrict__ W, int ldw,
                                        int nbase, int k0, int lane) {
  int n = nbase + (lane & 15);
  int kb = (lane & 16) ? 8 : 0;
  const float4* p = (const float4*)(W + (size_t)n * ldw + k0 + kb);
  float4 a0 = p[0], a1 = p[1], b0 = p[4], b1 = p[5];
  v16h r;
  r[0]=(_Float16)a0.x;  r[1]=(_Float16)a0.y;  r[2]=(_Float16)a0.z;  r[3]=(_Float16)a0.w;
  r[4]=(_Float16)a1.x;  r[5]=(_Float16)a1.y;  r[6]=(_Float16)a1.z;  r[7]=(_Float16)a1.w;
  r[8]=(_Float16)b0.x;  r[9]=(_Float16)b0.y;  r[10]=(_Float16)b0.z; r[11]=(_Float16)b0.w;
  r[12]=(_Float16)b1.x; r[13]=(_Float16)b1.y; r[14]=(_Float16)b1.z; r[15]=(_Float16)b1.w;
  return r;
}
// A tile from LDS f16 row-major; two ds_load_b128
__device__ __forceinline__ v16h ldA_s(const _Float16* sA, int lda, int k0,
                                      int lane) {
  int m = lane & 15;
  int kb = (lane & 16) ? 8 : 0;
  const _Float16* p = sA + m * lda + k0 + kb;
  return pack16(*(const v8h*)p, *(const v8h*)(p + 16));
}
// masked f16 operand (row pre-clamped); zeroes whole lane when invalid
__device__ __forceinline__ v16h ldMask(const _Float16* __restrict__ p, bool ok,
                                       int lane) {
  int kb = (lane & 16) ? 8 : 0;
  v8h lo = *(const v8h*)(p + kb);
  v8h hi = *(const v8h*)(p + kb + 16);
  v16h r;
#pragma unroll
  for (int e = 0; e < 8; ++e) {
    r[e]     = ok ? lo[e] : (_Float16)0.f;
    r[e + 8] = ok ? hi[e] : (_Float16)0.f;
  }
  return r;
}

__device__ __forceinline__ float gelu_exact(float x) {
  return 0.5f * x * (1.0f + erff(x * 0.70710678118654752f));
}

__device__ __forceinline__ float blk_sum128(float v, float* buf) {
  int t = threadIdx.x;
  buf[t] = v; __syncthreads();
#pragma unroll
  for (int s2 = 64; s2 > 0; s2 >>= 1) {
    if (t < s2) buf[t] += buf[t + s2];
    __syncthreads();
  }
  float r = buf[0]; __syncthreads();
  return r;
}

// ---- STFT + projection + LN + GELU + pos-embed -----------------------------
__global__ void k_stft(const float* __restrict__ x, const float* __restrict__ pw,
                       const float* __restrict__ pb, const float* __restrict__ pg,
                       const float* __restrict__ pbeta,
                       const float* __restrict__ pos, float* __restrict__ h) {
  int seg = blockIdx.x;  // = b*240 + n
  int tid = threadIdx.x;
  __shared__ float xs[SEGL];
  __shared__ float ct[64], st[64];
  __shared__ float mg[NFLAT];
  __shared__ float red[128];

  for (int i = tid; i < SEGL; i += 128) xs[i] = x[seg * SEGL + i];
  if (tid < 64) {
    float sv, cv;
    sincosf(6.2831853071795864769f * (float)tid / 64.0f, &sv, &cv);
    ct[tid] = cv; st[tid] = sv;
  }
  __syncthreads();

  for (int idx = tid; idx < NFLAT; idx += 128) {
    int f = idx >> 3, t = idx & 7;   // flat index = f*FRAMES + t
    float re = 0.f, im = 0.f;
    for (int n = 0; n < 64; ++n) {
      int j = t * 32 + n - 32;       // reflect pad by NFFT/2
      int jj = (j < 0) ? -j : ((j >= SEGL) ? (2 * (SEGL - 1) - j) : j);
      float v = xs[jj];
      int p = (f * n) & 63;
      re += v * ct[p];
      im -= v * st[p];
    }
    mg[idx] = sqrtf(re * re + im * im);
  }
  __syncthreads();

  int d = tid;
  float acc = pb[d];
  for (int i = 0; i < NFLAT; ++i) acc += mg[i] * pw[i * DMODEL + d];

  float mean = blk_sum128(acc, red) * (1.0f / 128.0f);
  float dv = acc - mean;
  float var = blk_sum128(dv * dv, red) * (1.0f / 128.0f);
  float y = dv * rsqrtf(var + 1e-5f) * pg[d] + pbeta[d];
  float g = gelu_exact(y);
  int b = seg / NTOKN, n = seg % NTOKN;
  h[(b * S_SEQ + 1 + n) * DMODEL + d] = g + pos[n * DMODEL + d];
}

__global__ void k_cls(const float* __restrict__ cls, float* __restrict__ h) {
  h[(blockIdx.x * S_SEQ) * DMODEL + threadIdx.x] = cls[threadIdx.x];
}

// ---- LayerNorm, writes f32 + f16 copies ------------------------------------
__global__ void k_ln(const float* __restrict__ x, float* __restrict__ y,
                     _Float16* __restrict__ y16,
                     const float* __restrict__ g, const float* __restrict__ b) {
  int r = blockIdx.x, d = threadIdx.x;
  __shared__ float red[128];
  float v = x[r * DMODEL + d];
  float mean = blk_sum128(v, red) * (1.0f / 128.0f);
  float dv = v - mean;
  float var = blk_sum128(dv * dv, red) * (1.0f / 128.0f);
  float o = dv * rsqrtf(var + 1e-5f) * g[d] + b[d];
  y[r * DMODEL + d] = o;
  y16[r * DMODEL + d] = (_Float16)o;
}

// ---- weight convert + transpose: src f32 (K,N) -> dst f16 (N,K) ------------
__global__ void k_prep(const float* __restrict__ src, _Float16* __restrict__ dst,
                       int K, int N) {
  int idx = blockIdx.x * 256 + threadIdx.x;
  if (idx >= K * N) return;
  int k = idx / N, n = idx % N;
  dst[n * K + k] = (_Float16)src[idx];
}

// ---- C f16 = A f16 @ W^T + bias (QKV projection) ---------------------------
__global__ void k_gemm_qkv(const _Float16* __restrict__ A, const float* __restrict__ W,
                           const float* __restrict__ bias, _Float16* __restrict__ C,
                           int Mtiles, int Ntiles, int K, int N) {
  int wv = (blockIdx.x << 3) + (threadIdx.x >> 5);
  int lane = threadIdx.x & 31;
  if (wv >= Mtiles * Ntiles) return;
  int tm = wv / Ntiles, tn = wv % Ntiles;
  v8f acc = {};
  for (int k0 = 0; k0 < K; k0 += 32)
    acc = wmma16(ldA_h(A, K, tm * 16, k0, lane),
                 ldB_wT4(W, K, tn * 16, k0, lane), acc);
  int n = tn * 16 + (lane & 15);
  float bn = bias[n];
#pragma unroll
  for (int i = 0; i < 8; ++i) {
    int m = tm * 16 + i + ((lane & 16) ? 8 : 0);
    C[(size_t)m * N + n] = (_Float16)(acc[i] + bn);
  }
}

// ---- H f32 += A f16 @ W^T + bias (attention out-proj + residual) -----------
__global__ void k_gemm_out(const _Float16* __restrict__ A, const float* __restrict__ W,
                           const float* __restrict__ bias, float* __restrict__ H,
                           int Mtiles, int Ntiles, int K, int N) {
  int wv = (blockIdx.x << 3) + (threadIdx.x >> 5);
  int lane = threadIdx.x & 31;
  if (wv >= Mtiles * Ntiles) return;
  int tm = wv / Ntiles, tn = wv % Ntiles;
  v8f acc = {};
  for (int k0 = 0; k0 < K; k0 += 32)
    acc = wmma16(ldA_h(A, K, tm * 16, k0, lane),
                 ldB_wT4(W, K, tn * 16, k0, lane), acc);
  int n = tn * 16 + (lane & 15);
  float bn = bias[n];
#pragma unroll
  for (int i = 0; i < 8; ++i) {
    int m = tm * 16 + i + ((lane & 16) ? 8 : 0);
    size_t id = (size_t)m * N + n;
    H[id] = H[id] + acc[i] + bn;
  }
}

// ---- V transpose: qkv f16 -> vt[(b*4+h)*32+dh][s] f16, zero pad s>=241 -----
__global__ void k_vt(const _Float16* __restrict__ qkv, _Float16* __restrict__ vt) {
  int idx = blockIdx.x * 256 + threadIdx.x;
  int s  = idx & 255;
  int dh = (idx >> 8) & 31;
  int hh = (idx >> 13) & 3;
  int b  = idx >> 15;
  _Float16 v = (_Float16)0.f;
  if (s < S_SEQ) v = qkv[(size_t)(b * S_SEQ + s) * 384 + 256 + hh * DHEAD + dh];
  vt[idx] = v;
}

// ---- Attention: one wave per (b, head, 16-query tile) ----------------------
__global__ void k_attn(const _Float16* __restrict__ qkv,
                       const _Float16* __restrict__ vt,
                       _Float16* __restrict__ o) {
  const float scale = 0.17677669529663689f;  // 1/sqrt(32)
  int bid = blockIdx.x;
  int qt = bid & 15;
  int hh = (bid >> 4) & 3;
  int b  = bid >> 6;
  int lane = threadIdx.x;
  __shared__ float    sc[16 * 256];
  __shared__ _Float16 at[16 * 256];
  int q0 = qt * 16;

  v16h a;  // Q tile 16x32
  {
    int m = lane & 15;
    int qi = q0 + m;
    int qc = qi < S_SEQ ? qi : S_SEQ - 1;
    a = ldMask(qkv + (size_t)(b * S_SEQ + qc) * 384 + hh * DHEAD, qi < S_SEQ, lane);
  }
  for (int kt = 0; kt < 16; ++kt) {
    int kj0 = kt * 16;
    int n = lane & 15;
    int kj = kj0 + n;
    int kc = kj < S_SEQ ? kj : S_SEQ - 1;
    v16h bb = ldMask(qkv + (size_t)(b * S_SEQ + kc) * 384 + 128 + hh * DHEAD,
                     kj < S_SEQ, lane);
    v8f acc = {};
    acc = wmma16(a, bb, acc);
#pragma unroll
    for (int i = 0; i < 8; ++i) {
      int m = i + ((lane & 16) ? 8 : 0);
      sc[m * 256 + kj] = (kj < S_SEQ) ? acc[i] * scale : -1e30f;
    }
  }
  __syncthreads();
  if (lane < 16) {
    int m = lane;
    float mx = -1e30f;
    for (int j = 0; j < S_SEQ; ++j) mx = fmaxf(mx, sc[m * 256 + j]);
    float sm = 0.f;
    for (int j = 0; j < S_SEQ; ++j) {
      float ev = expf(sc[m * 256 + j] - mx);
      sc[m * 256 + j] = ev; sm += ev;
    }
    float inv = 1.f / sm;
    for (int j = 0; j < S_SEQ; ++j) at[m * 256 + j] = (_Float16)(sc[m * 256 + j] * inv);
    for (int j = S_SEQ; j < 256; ++j) at[m * 256 + j] = (_Float16)0.f;
  }
  __syncthreads();
  const _Float16* vbase = vt + (size_t)(b * NHEAD + hh) * DHEAD * 256;
  for (int nt = 0; nt < 2; ++nt) {
    v8f acc = {};
    for (int k0 = 0; k0 < 256; k0 += 32)
      acc = wmma16(ldA_s(at, 256, k0, lane), ldB_hT(vbase, 256, nt * 16, k0, lane),
                   acc);
    int n = lane & 15;
#pragma unroll
    for (int i = 0; i < 8; ++i) {
      int m = i + ((lane & 16) ? 8 : 0);
      int qi = q0 + m;
      if (qi < S_SEQ)
        o[(size_t)(b * S_SEQ + qi) * DMODEL + hh * DHEAD + nt * 16 + n] =
            (_Float16)acc[i];
    }
  }
}

// ---- Fused MoE: TDM tile stage + gate + universal FFN + active experts -----
__global__ void k_moe(const float* __restrict__ h2, const _Float16* __restrict__ h2h,
                      float* __restrict__ h, const int* __restrict__ task_ids,
                      const float* __restrict__ temb,
                      const float* __restrict__ gw, const float* __restrict__ gb,
                      const _Float16* __restrict__ ew1T, const float* __restrict__ eb1,
                      const _Float16* __restrict__ ew2T, const float* __restrict__ eb2,
                      const _Float16* __restrict__ uw1T, const float* __restrict__ ub1,
                      const _Float16* __restrict__ uw2T, const float* __restrict__ ub2,
                      float* __restrict__ router, int l) {
  int r0 = blockIdx.x * 16;
  int tid = threadIdx.x;
  int lane = tid & 31, wv = tid >> 5;
  __shared__ _Float16 sA[16 * DMODEL];
  __shared__ _Float16 sHid[16 * DFFN];
  __shared__ float sUo[16 * DMODEL];
  __shared__ float sGate[16 * NEXP];
  __shared__ float sOmega[16];
  __shared__ int   sAct[NEXP];

  // stage the 16x128 f16 activation tile into LDS via the Tensor Data Mover
#if __has_builtin(__builtin_amdgcn_tensor_load_to_lds)
  if (wv == 0) {
    unsigned long long ga =
        (unsigned long long)(size_t)(h2h + (size_t)r0 * DMODEL);
    unsigned ldsa = (unsigned)(size_t)(void*)sA;  // flat low 32 = LDS offset
    v4u g0; g0[0] = 1u;                       // count=1, user descriptor
    g0[1] = ldsa;                             // lds_addr
    g0[2] = (unsigned)(ga & 0xFFFFFFFFu);     // global_addr[31:0]
    g0[3] = (unsigned)((ga >> 32) & 0x01FFFFFFull) | 0x80000000u;  // type=2
    v8i g1;
    g1[0] = (int)(1u << 16);        // data_size=1 (2-byte elements)
    g1[1] = (int)(DMODEL << 16);    // tensor_dim0 = 128
    g1[2] = (int)(16 << 16);        // tensor_dim1 = 16
    g1[3] = (int)(DMODEL << 16);    // tile_dim0 = 128
    g1[4] = 16;                     // tile_dim1 = 16
    g1[5] = DMODEL;                 // tensor_dim0_stride = 128
    g1[6] = 0; g1[7] = 0;
    v4i gz4; gz4[0] = 0; gz4[1] = 0; gz4[2] = 0; gz4[3] = 0;
    v8i gz8;
#pragma unroll
    for (int q = 0; q < 8; ++q) gz8[q] = 0;
    __builtin_amdgcn_tensor_load_to_lds(g0, g1, gz4, gz4, gz8, 0);
    __builtin_amdgcn_s_wait_tensorcnt(0);
  }
#else
  for (int i = tid; i < 16 * DMODEL; i += 256)
    sA[i] = h2h[(size_t)r0 * DMODEL + i];
#endif
  __syncthreads();

  // gate logits + router output
  if (tid < 128) {
    int row = tid >> 3, e = tid & 7;
    int r = r0 + row, b = r / S_SEQ, s = r % S_SEQ;
    int t = task_ids[b];
    const float* gwp = gw + e * 256;
    const float* tv = temb + t * DMODEL;
    const float* hp = h2 + (size_t)r * DMODEL;
    float acc = gb[e];
    for (int d2 = 0; d2 < DMODEL; ++d2)
      acc += hp[d2] * gwp[d2] + tv[d2] * gwp[128 + d2];
    sGate[row * 8 + e] = acc;  // logits (replaced by gates below)
    router[((size_t)(b * NLAYER + l) * S_SEQ + s) * NEXP + e] = acc;
  }
  __syncthreads();
  if (tid < 16) {
    float v0 = -1e30f, v1 = -1e30f; int i0 = 0, i1 = 0;
    for (int e = 0; e < 8; ++e) {
      float v = sGate[tid * 8 + e];
      if (v > v0)      { v1 = v0; i1 = i0; v0 = v; i0 = e; }
      else if (v > v1) { v1 = v; i1 = e; }
    }
    float g0 = 1.f / (1.f + expf(v1 - v0));
    float g1 = 1.f - g0;
    for (int e = 0; e < 8; ++e) sGate[tid * 8 + e] = 0.f;
    sGate[tid * 8 + i0] = g0;
    sGate[tid * 8 + i1] = g1;
    sOmega[tid] = 1.f - g0;
  }
  __syncthreads();
  if (tid < 8) {
    int any = 0;
    for (int row = 0; row < 16; ++row) any |= (sGate[row * 8 + tid] != 0.f);
    sAct[tid] = any;
  }

  // universal FFN hidden: wave w owns cols [w*64, w*64+64)
  for (int j = 0; j < 4; ++j) {
    int n0 = wv * 64 + j * 16;
    v8f acc = {};
    for (int k0 = 0; k0 < DMODEL; k0 += 32)
      acc = wmma16(ldA_s(sA, DMODEL, k0, lane),
                   ldB_hT(uw1T, DMODEL, n0, k0, lane), acc);
    int n = n0 + (lane & 15);
#pragma unroll
    for (int i = 0; i < 8; ++i) {
      int m = i + ((lane & 16) ? 8 : 0);
      sHid[m * DFFN + n] = (_Float16)gelu_exact(acc[i] + ub1[n]);
    }
  }
  __syncthreads();
  {  // universal FFN out: wave w owns cols [w*16, w*16+16)
    int n0 = wv * 16;
    v8f acc = {};
    for (int k0 = 0; k0 < DFFN; k0 += 32)
      acc = wmma16(ldA_s(sHid, DFFN, k0, lane),
                   ldB_hT(uw2T, DFFN, n0, k0, lane), acc);
    int n = n0 + (lane & 15);
#pragma unroll
    for (int i = 0; i < 8; ++i) {
      int m = i + ((lane & 16) ? 8 : 0);
      sUo[m * DMODEL + n] = acc[i] + ub2[n];
    }
  }

  v8f tso = {};
  for (int e = 0; e < NEXP; ++e) {
    __syncthreads();          // protects sHid reuse & sAct/sUo first-iter deps
    if (!sAct[e]) continue;   // uniform across block
    const _Float16* w1 = ew1T + (size_t)e * DFFN * DMODEL;  // (512,128) f16
    const float* b1 = eb1 + e * DFFN;
    for (int j = 0; j < 4; ++j) {
      int n0 = wv * 64 + j * 16;
      v8f acc = {};
      for (int k0 = 0; k0 < DMODEL; k0 += 32)
        acc = wmma16(ldA_s(sA, DMODEL, k0, lane),
                     ldB_hT(w1, DMODEL, n0, k0, lane), acc);
      int n = n0 + (lane & 15);
#pragma unroll
      for (int i = 0; i < 8; ++i) {
        int m = i + ((lane & 16) ? 8 : 0);
        sHid[m * DFFN + n] = (_Float16)gelu_exact(acc[i] + b1[n]);
      }
    }
    __syncthreads();
    const _Float16* w2 = ew2T + (size_t)e * DMODEL * DFFN;  // (128,512) f16
    const float* b2 = eb2 + e * DMODEL;
    int n0 = wv * 16;
    v8f acc = {};
    for (int k0 = 0; k0 < DFFN; k0 += 32)
      acc = wmma16(ldA_s(sHid, DFFN, k0, lane),
                   ldB_hT(w2, DFFN, n0, k0, lane), acc);
    int n = n0 + (lane & 15);
#pragma unroll
    for (int i = 0; i < 8; ++i) {
      int m = i + ((lane & 16) ? 8 : 0);
      tso[i] += sGate[m * 8 + e] * (acc[i] + b2[n]);
    }
  }
  __syncthreads();
  {  // h += tso + omega * uo
    int col = wv * 16 + (lane & 15);
#pragma unroll
    for (int i = 0; i < 8; ++i) {
      int m = i + ((lane & 16) ? 8 : 0);
      int r = r0 + m;
      h[(size_t)r * DMODEL + col] += tso[i] + sOmega[m] * sUo[m * DMODEL + col];
    }
  }
}

// ---- Final LN on CLS row + task head ---------------------------------------
__global__ void k_head(const float* __restrict__ h, const int* __restrict__ task_ids,
                       const float* __restrict__ fg, const float* __restrict__ fb,
                       const float* __restrict__ hw, const float* __restrict__ hb,
                       float* __restrict__ out) {
  int b = blockIdx.x, d = threadIdx.x;
  __shared__ float red[128];
  float v = h[(size_t)b * S_SEQ * DMODEL + d];
  float mean = blk_sum128(v, red) * (1.0f / 128.0f);
  float dv = v - mean;
  float var = blk_sum128(dv * dv, red) * (1.0f / 128.0f);
  float cls = dv * rsqrtf(var + 1e-5f) * fg[d] + fb[d];
  int t = task_ids[b];
  float o0 = blk_sum128(cls * hw[(t * DMODEL + d) * 2 + 0], red);
  float o1 = blk_sum128(cls * hw[(t * DMODEL + d) * 2 + 1], red);
  if (d == 0) {
    out[b * 2 + 0] = o0 + hb[t * 2 + 0];
    out[b * 2 + 1] = o1 + hb[t * 2 + 1];
  }
}

// ---- host launcher ---------------------------------------------------------
extern "C" void kernel_launch(void* const* d_in, const int* in_sizes, int n_in,
                              void* d_out, int out_size, void* d_ws, size_t ws_size,
                              hipStream_t stream) {
  (void)in_sizes; (void)n_in; (void)out_size; (void)ws_size;
  const float* x         = (const float*)d_in[0];
  const int*   task_ids  = (const int*)  d_in[1];
  const float* proj_w    = (const float*)d_in[2];
  const float* proj_b    = (const float*)d_in[3];
  const float* proj_g    = (const float*)d_in[4];
  const float* proj_beta = (const float*)d_in[5];
  const float* pos_embed = (const float*)d_in[6];
  const float* cls_token = (const float*)d_in[7];
  const float* ln1_g     = (const float*)d_in[8];
  const float* ln1_b     = (const float*)d_in[9];
  const float* attn_in_w = (const float*)d_in[10];
  const float* attn_in_b = (const float*)d_in[11];
  const float* attn_out_w= (const float*)d_in[12];
  const float* attn_out_b= (const float*)d_in[13];
  const float* ln2_g     = (const float*)d_in[14];
  const float* ln2_b     = (const float*)d_in[15];
  const float* task_embed= (const float*)d_in[16];
  const float* gate_w    = (const float*)d_in[17];
  const float* gate_b    = (const float*)d_in[18];
  const float* exp_w1    = (const float*)d_in[19];
  const float* exp_b1    = (const float*)d_in[20];
  const float* exp_w2    = (const float*)d_in[21];
  const float* exp_b2    = (const float*)d_in[22];
  const float* uni_w1    = (const float*)d_in[23];
  const float* uni_b1    = (const float*)d_in[24];
  const float* uni_w2    = (const float*)d_in[25];
  const float* uni_b2    = (const float*)d_in[26];
  const float* fin_g     = (const float*)d_in[27];
  const float* fin_b     = (const float*)d_in[28];
  const float* head_w    = (const float*)d_in[29];
  const float* head_b    = (const float*)d_in[30];
  float* out = (float*)d_out;

  const size_t ND  = (size_t)NROWS * DMODEL;        // 3,948,544
  const size_t W65 = (size_t)DMODEL * DFFN;         // 65,536
  float* h   = (float*)d_ws;
  float* h2  = h + ND;
  _Float16* h2h  = (_Float16*)(h2 + ND);
  _Float16* obh  = h2h + ND;
  _Float16* qkvh = obh + ND;
  _Float16* vt   = qkvh + (size_t)NROWS * 384;
  _Float16* wt   = vt + (size_t)NBATCH * NHEAD * DHEAD * 256;
  // per-layer f16 weight region: uw1T | uw2T | ew1T[8] | ew2T[8]
  const size_t WPL = 18 * W65;

  // convert+transpose all FFN weights to f16 (N,K)
  for (int l = 0; l < NLAYER; ++l) {
    _Float16* wl = wt + (size_t)l * WPL;
    k_prep<<<256, 256, 0, stream>>>(uni_w1 + (size_t)l * W65, wl, DMODEL, DFFN);
    k_prep<<<256, 256, 0, stream>>>(uni_w2 + (size_t)l * W65, wl + W65, DFFN, DMODEL);
    for (int e = 0; e < NEXP; ++e) {
      k_prep<<<256, 256, 0, stream>>>(exp_w1 + ((size_t)l * NEXP + e) * W65,
                                      wl + (2 + e) * W65, DMODEL, DFFN);
      k_prep<<<256, 256, 0, stream>>>(exp_w2 + ((size_t)l * NEXP + e) * W65,
                                      wl + (10 + e) * W65, DFFN, DMODEL);
    }
  }

  k_stft<<<NBATCH * NTOKN, 128, 0, stream>>>(x, proj_w, proj_b, proj_g,
                                             proj_beta, pos_embed, h);
  k_cls<<<NBATCH, 128, 0, stream>>>(cls_token, h);

  const int Mt = NROWS / 16;  // 1928, exact
  for (int l = 0; l < NLAYER; ++l) {
    _Float16* wl = wt + (size_t)l * WPL;
    k_ln<<<NROWS, 128, 0, stream>>>(h, h2, h2h, ln1_g + l * DMODEL,
                                    ln1_b + l * DMODEL);
    k_gemm_qkv<<<(Mt * 24 + 7) / 8, 256, 0, stream>>>(
        h2h, attn_in_w + (size_t)l * 384 * DMODEL, attn_in_b + l * 384, qkvh,
        Mt, 24, DMODEL, 384);
    k_vt<<<(NBATCH * NHEAD * DHEAD * 256) / 256, 256, 0, stream>>>(qkvh, vt);
    k_attn<<<NBATCH * NHEAD * 16, 32, 0, stream>>>(qkvh, vt, obh);
    k_gemm_out<<<(Mt * 8 + 7) / 8, 256, 0, stream>>>(
        obh, attn_out_w + (size_t)l * DMODEL * DMODEL, attn_out_b + l * DMODEL,
        h, Mt, 8, DMODEL, DMODEL);
    k_ln<<<NROWS, 128, 0, stream>>>(h, h2, h2h, ln2_g + l * DMODEL,
                                    ln2_b + l * DMODEL);
    k_moe<<<Mt, 256, 0, stream>>>(
        h2, h2h, h, task_ids, task_embed + (size_t)l * 5 * DMODEL,
        gate_w + (size_t)l * NEXP * 256, gate_b + l * NEXP,
        wl + 2 * W65,  exp_b1 + (size_t)l * NEXP * DFFN,
        wl + 10 * W65, exp_b2 + (size_t)l * NEXP * DMODEL,
        wl,            uni_b1 + (size_t)l * DFFN,
        wl + W65,      uni_b2 + (size_t)l * DMODEL,
        out + 2 * NBATCH, l);
  }
  k_head<<<NBATCH, 128, 0, stream>>>(h, task_ids, fin_g, fin_b, head_w, head_b,
                                     out);
}
